// Attention_42949673732
// MI455X (gfx1250) — matrix operand order repaired and checked by reading, not compiled, not run on hardware
//
#include <hip/hip_runtime.h>
#include <hip/hip_bf16.h>

typedef __bf16 bf16_t;
typedef __attribute__((ext_vector_type(16))) __bf16 v16bf;
typedef __attribute__((ext_vector_type(8)))  __bf16 v8bf;
typedef __attribute__((ext_vector_type(8)))  float  v8f;
typedef __attribute__((ext_vector_type(4)))  unsigned int u32x4;
typedef __attribute__((ext_vector_type(8)))  int i32x8;
typedef __attribute__((ext_vector_type(4)))  int i32x4;

#define D_MODEL 2048
#define BATCH   2
#define SEQ     2048
#define NHEADS  32
#define DHEAD   64
#define MROWS   (BATCH*SEQ)   // 4096

union AV { v16bf v; v8bf h[2]; bf16_t e[16]; };

// ---------------- fp32 -> bf16 conversion ----------------
__global__ void cvt_f32_bf16(const float* __restrict__ in, bf16_t* __restrict__ out, int n) {
    int i = blockIdx.x * blockDim.x + threadIdx.x;
    if (i < n) out[i] = (bf16_t)in[i];
}

// ---------------- WMMA GEMM: C[m,n] = sum_k A[m,k]*W[n,k] + bias[n] ----------------
// A: [M,K] bf16 row-major.  W: [N,K] bf16 row-major (torch Linear weight).
// Block = 256 thr = 8 waves; block tile M=64 x N=256; each wave: 32x64 (2x4 C tiles).
template<bool OUT_F32>
__global__ void __launch_bounds__(256)
gemm_bf16_wmma(const bf16_t* __restrict__ A,
               const bf16_t* __restrict__ W,
               const float*  __restrict__ bias,
               bf16_t* __restrict__ outb,
               float*  __restrict__ outf,
               int M, int N, int K)
{
    const int lane = threadIdx.x & 31;
    const int wave = threadIdx.x >> 5;
    const int hlf  = lane >> 4;        // 0: lanes 0-15, 1: lanes 16-31
    const int l15  = lane & 15;
    const int m0 = blockIdx.y * 64  + (wave >> 2) * 32;
    const int n0 = blockIdx.x * 256 + (wave & 3) * 64;

    v8f acc[2][4];
#pragma unroll
    for (int mi = 0; mi < 2; ++mi)
#pragma unroll
        for (int t = 0; t < 4; ++t) acc[mi][t] = 0.f;

    const bf16_t* arow0 = A + (size_t)(m0 + l15) * K;
    const bf16_t* arow1 = arow0 + (size_t)16 * K;
    const bf16_t* wr[4];
#pragma unroll
    for (int t = 0; t < 4; ++t)
        wr[t] = W + (size_t)(n0 + t * 16 + l15) * K + hlf * 16;

    for (int k = 0; k < K; k += 32) {
        // A 16x32 bf16 layout: lane row = l15; element i <-> k_local = hlf*16 + i
        AV a0, a1;
        a0.h[0] = *(const v8bf*)(arow0 + k + hlf * 16);
        a0.h[1] = *(const v8bf*)(arow0 + k + hlf * 16 + 8);
        a1.h[0] = *(const v8bf*)(arow1 + k + hlf * 16);
        a1.h[1] = *(const v8bf*)(arow1 + k + hlf * 16 + 8);
#pragma unroll
        for (int t = 0; t < 4; ++t) {
            // B 32x16 bf16 layout: lane col = l15; element i <-> k_local = hlf*16 + i
            AV b;
            b.h[0] = *(const v8bf*)(wr[t] + k);
            b.h[1] = *(const v8bf*)(wr[t] + k + 8);
            acc[0][t] = __builtin_amdgcn_wmma_f32_16x16x32_bf16(
                false, a0.v, false, b.v, (short)0, acc[0][t], false, false);
            acc[1][t] = __builtin_amdgcn_wmma_f32_16x16x32_bf16(
                false, a1.v, false, b.v, (short)0, acc[1][t], false, false);
        }
    }

#pragma unroll
    for (int mi = 0; mi < 2; ++mi)
#pragma unroll
        for (int t = 0; t < 4; ++t) {
            const int n = n0 + t * 16 + l15;
            const float bv = bias ? bias[n] : 0.f;
#pragma unroll
            for (int r = 0; r < 8; ++r) {
                const int m = m0 + mi * 16 + r + hlf * 8;   // C layout: M = vgpr + 8*(lane>=16)
                const float v = acc[mi][t][r] + bv;
                if (OUT_F32) outf[(size_t)m * N + n] = v;
                else         outb[(size_t)m * N + n] = (bf16_t)v;
            }
        }
}

// ---------------- RoPE in place on [B*S, H*Dh] bf16 ----------------
__global__ void rope_inplace(bf16_t* __restrict__ X) {
    int idx = blockIdx.x * blockDim.x + threadIdx.x;     // B*S*H*32 threads
    int d = idx & 31;
    int h = (idx >> 5) & (NHEADS - 1);
    int s = (idx >> 10) & (SEQ - 1);
    int b = idx >> 21;
    size_t base = ((size_t)(b * SEQ + s)) * D_MODEL + h * DHEAD;
    float inv = exp2f(-(float)d * (13.287712379549449f / 32.f));  // 10000^(-d/32)
    float ang = (float)s * inv;
    float sn, c;
    __sincosf(ang, &sn, &c);
    float x0 = (float)X[base + d];
    float x1 = (float)X[base + d + 32];
    X[base + d]      = (bf16_t)(x0 * c - x1 * sn);
    X[base + d + 32] = (bf16_t)(x1 * c + x0 * sn);
}

// ---------------- sigmoid in place ----------------
__global__ void sigmoid_inplace(bf16_t* __restrict__ V, int n) {
    int i = blockIdx.x * blockDim.x + threadIdx.x;
    if (i < n) {
        float v = (float)V[i];
        V[i] = (bf16_t)(1.f / (1.f + __expf(-v)));
    }
}

// ---------------- flash attention: one wave per 16-query tile ----------------
// Q,K,V: bf16 [B*S, H*Dh]; Y: bf16 [B*S, H*Dh]. Online softmax over 32-key blocks.
// V tiles (32 keys x 64 dh) are DMA'd to LDS by the Tensor Data Mover, double-buffered,
// synchronized with TENSORcnt.
__global__ void __launch_bounds__(32)
attn_fwd(const bf16_t* __restrict__ Q,
         const bf16_t* __restrict__ K,
         const bf16_t* __restrict__ V,
         bf16_t* __restrict__ Y)
{
    __shared__ __align__(16) bf16_t vstage[2][32][64];   // TDM double buffer (8 KB)
    __shared__ __align__(16) bf16_t psh[16][32];         // P tile C-layout -> A-layout staging

    const int lane = threadIdx.x & 31;
    const int hlf = lane >> 4, l15 = lane & 15;
    const int bx = blockIdx.x;
    const int qt = bx & (SEQ / 16 - 1);
    const int hd = (bx >> 7) & (NHEADS - 1);
    const int b  = bx >> 12;
    const int qbase = qt * 16;
    const size_t rowbase = (size_t)b * SEQ;
    const int col0 = hd * DHEAD;

    // TDM descriptor: 2D tile, 64 elems/row (Dh) x 32 rows (keys), 2B elems,
    // row stride = D_MODEL elements. D# per CDNA5 ISA 8.3/8.4.
    auto tdm_load_v = [&](int key0, int buf) {
        unsigned long long ga =
            (unsigned long long)(const void*)(V + (rowbase + key0) * D_MODEL + col0);
        unsigned int la = (unsigned int)(unsigned long long)&vstage[buf][0][0];
        u32x4 g0;
        g0[0] = 1u;                                     // count=1 (valid), user mode
        g0[1] = la;                                     // lds_addr
        g0[2] = (unsigned int)ga;                       // global_addr[31:0]
        g0[3] = (unsigned int)((ga >> 32) & 0x01FFFFFFu) | (2u << 30);  // addr[56:32] | type=2
        i32x8 g1;
        g1[0] = (int)(1u << 16);                        // data_size=1 (2 bytes)
        g1[1] = (int)(64u << 16);                       // tensor_dim0 = 64 (lo16 in [31:16])
        g1[2] = (int)(32u << 16);                       // tensor_dim0 hi=0 | tensor_dim1=32 lo16
        g1[3] = (int)(64u << 16);                       // tensor_dim1 hi=0 | tile_dim0 = 64
        g1[4] = (int)32u;                               // tile_dim1 = 32, tile_dim2 = 0
        g1[5] = (int)2048u;                             // tensor_dim0_stride lo32 = D_MODEL
        g1[6] = 0;                                      // stride0 hi | stride1 lo
        g1[7] = 0;                                      // stride1 hi
        i32x4 z4;
        z4[0] = 0; z4[1] = 0; z4[2] = 0; z4[3] = 0;
        i32x8 z8;
#pragma unroll
        for (int i = 0; i < 8; ++i) z8[i] = 0;
        __builtin_amdgcn_tensor_load_to_lds(g0, g1, z4, z4, z8, 0);
    };

    // Q as two 16x32 A-matrices (Dh = 64 = 2 chunks of K=32)
    AV aq[2];
    {
        const bf16_t* qrow = Q + (rowbase + qbase + l15) * D_MODEL + col0;
#pragma unroll
        for (int c = 0; c < 2; ++c) {
            aq[c].h[0] = *(const v8bf*)(qrow + c * 32 + hlf * 16);
            aq[c].h[1] = *(const v8bf*)(qrow + c * 32 + hlf * 16 + 8);
        }
    }

    float mrun[8], lrun[8];
#pragma unroll
    for (int r = 0; r < 8; ++r) { mrun[r] = -1e30f; lrun[r] = 0.f; }
    v8f o[4];
#pragma unroll
    for (int t = 0; t < 4; ++t) o[t] = 0.f;

    const float scale = 0.125f;                 // 1/sqrt(64)
    const int nkb = (qbase + 47) >> 5;          // 32-key blocks covering keys <= qbase+15

    tdm_load_v(0, 0);                           // prefetch first V tile

    for (int j = 0; j < nkb; ++j) {
        const int key0 = j * 32;
        const int cur = j & 1;
        const bool have_next = (j + 1 < nkb);   // uniform across wave
        if (have_next) tdm_load_v(key0 + 32, cur ^ 1);   // async prefetch next

        // S = Q @ K^T * scale : two 16x16 score tiles (keys key0..+15, key0+16..+31)
        v8f s[2];
#pragma unroll
        for (int st = 0; st < 2; ++st) {
            s[st] = 0.f;
            const bf16_t* krow = K + (rowbase + key0 + st * 16 + l15) * D_MODEL + col0 + hlf * 16;
#pragma unroll
            for (int c = 0; c < 2; ++c) {
                AV bk;
                bk.h[0] = *(const v8bf*)(krow + c * 32);
                bk.h[1] = *(const v8bf*)(krow + c * 32 + 8);
                s[st] = __builtin_amdgcn_wmma_f32_16x16x32_bf16(
                    false, aq[c].v, false, bk.v, (short)0, s[st], false, false);
            }
        }

        const bool need_mask = (key0 + 31) > qbase;   // uniform across wave
        float mx[8];
#pragma unroll
        for (int r = 0; r < 8; ++r) {
            float s0 = s[0][r] * scale, s1 = s[1][r] * scale;
            if (need_mask) {
                int q = qbase + r + hlf * 8;
                if (key0 + l15 > q)      s0 = -1e30f;
                if (key0 + 16 + l15 > q) s1 = -1e30f;
            }
            s[0][r] = s0; s[1][r] = s1;
            float m = fmaxf(s0, s1);
#pragma unroll
            for (int off = 8; off >= 1; off >>= 1)
                m = fmaxf(m, __shfl_xor(m, off, 16));   // reduce across N (16 lanes)
            mx[r] = m;
        }

#pragma unroll
        for (int r = 0; r < 8; ++r) {
            float mnew  = fmaxf(mrun[r], mx[r]);
            float alpha = __expf(mrun[r] - mnew);
            mrun[r] = mnew;
            float p0 = __expf(s[0][r] - mnew);
            float p1 = __expf(s[1][r] - mnew);
            float rs = p0 + p1;
#pragma unroll
            for (int off = 8; off >= 1; off >>= 1)
                rs += __shfl_xor(rs, off, 16);
            lrun[r] = lrun[r] * alpha + rs;
#pragma unroll
            for (int t = 0; t < 4; ++t) o[t][r] *= alpha;
            psh[r + hlf * 8][l15]      = (bf16_t)p0;   // P tile in (row, key) order
            psh[r + hlf * 8][16 + l15] = (bf16_t)p1;
        }
        __syncthreads();   // single-wave WG: barrier is S_NOP, but orders DS ops

        // reload P in A-matrix 16x32 layout
        AV ap;
        ap.h[0] = *(const v8bf*)&psh[l15][hlf * 16];
        ap.h[1] = *(const v8bf*)&psh[l15][hlf * 16 + 8];
        __syncthreads();

        // wait for current V tile DMA (allow the prefetch to remain in flight);
        // s_wait_tensorcnt requires an immediate operand -> uniform branch
        if (have_next) __builtin_amdgcn_s_wait_tensorcnt(1);
        else           __builtin_amdgcn_s_wait_tensorcnt(0);

        // O += P @ V : 4 output tiles across Dh; B from LDS (B-layout: k = hlf*16+i)
#pragma unroll
        for (int t = 0; t < 4; ++t) {
            AV bv;
#pragma unroll
            for (int i = 0; i < 16; ++i)
                bv.e[i] = vstage[cur][hlf * 16 + i][t * 16 + l15];
            o[t] = __builtin_amdgcn_wmma_f32_16x16x32_bf16(
                false, ap.v, false, bv.v, (short)0, o[t], false, false);
        }
    }

    float inv[8];
#pragma unroll
    for (int r = 0; r < 8; ++r) inv[r] = 1.f / lrun[r];
#pragma unroll
    for (int t = 0; t < 4; ++t)
#pragma unroll
        for (int r = 0; r < 8; ++r) {
            int m = qbase + r + hlf * 8;
            Y[(rowbase + m) * D_MODEL + col0 + t * 16 + l15] = (bf16_t)(o[t][r] * inv[r]);
        }
}

// ---------------- launch ----------------
extern "C" void kernel_launch(void* const* d_in, const int* in_sizes, int n_in,
                              void* d_out, int out_size, void* d_ws, size_t ws_size,
                              hipStream_t stream)
{
    (void)in_sizes; (void)n_in; (void)out_size; (void)ws_size;
    const float* x  = (const float*)d_in[0];
    const float* Wq = (const float*)d_in[1];
    const float* bq = (const float*)d_in[2];
    const float* Wk = (const float*)d_in[3];
    const float* bk = (const float*)d_in[4];
    const float* Wv = (const float*)d_in[5];
    const float* bv = (const float*)d_in[6];
    const float* Wo = (const float*)d_in[7];
    const float* bo = (const float*)d_in[8];
    float* out = (float*)d_out;

    const size_t SZ_X = (size_t)MROWS * D_MODEL;     // 8M elements
    const size_t SZ_W = (size_t)D_MODEL * D_MODEL;   // 4M elements
    bf16_t* xb  = (bf16_t*)d_ws;          // 16 MB
    bf16_t* wqb = xb  + SZ_X;             // 8 MB each
    bf16_t* wkb = wqb + SZ_W;
    bf16_t* wvb = wkb + SZ_W;
    bf16_t* wob = wvb + SZ_W;
    bf16_t* Qb  = wob + SZ_W;             // 16 MB each
    bf16_t* Kb  = Qb  + SZ_X;
    bf16_t* Vb  = Kb  + SZ_X;
    bf16_t* Yb  = Vb  + SZ_X;             // total ~112 MB

    cvt_f32_bf16<<<dim3((unsigned)((SZ_X + 255) / 256)), dim3(256), 0, stream>>>(x,  xb,  (int)SZ_X);
    cvt_f32_bf16<<<dim3((unsigned)((SZ_W + 255) / 256)), dim3(256), 0, stream>>>(Wq, wqb, (int)SZ_W);
    cvt_f32_bf16<<<dim3((unsigned)((SZ_W + 255) / 256)), dim3(256), 0, stream>>>(Wk, wkb, (int)SZ_W);
    cvt_f32_bf16<<<dim3((unsigned)((SZ_W + 255) / 256)), dim3(256), 0, stream>>>(Wv, wvb, (int)SZ_W);
    cvt_f32_bf16<<<dim3((unsigned)((SZ_W + 255) / 256)), dim3(256), 0, stream>>>(Wo, wob, (int)SZ_W);

    dim3 gg(D_MODEL / 256, MROWS / 64), gb(256);
    gemm_bf16_wmma<false><<<gg, gb, 0, stream>>>(xb, wqb, bq, Qb, nullptr, MROWS, D_MODEL, D_MODEL);
    gemm_bf16_wmma<false><<<gg, gb, 0, stream>>>(xb, wkb, bk, Kb, nullptr, MROWS, D_MODEL, D_MODEL);
    gemm_bf16_wmma<false><<<gg, gb, 0, stream>>>(xb, wvb, bv, Vb, nullptr, MROWS, D_MODEL, D_MODEL);

    const int nrope = BATCH * SEQ * NHEADS * 32;     // 2^22
    rope_inplace<<<dim3(nrope / 256), dim3(256), 0, stream>>>(Qb);
    rope_inplace<<<dim3(nrope / 256), dim3(256), 0, stream>>>(Kb);
    sigmoid_inplace<<<dim3((unsigned)(SZ_X / 256)), dim3(256), 0, stream>>>(Vb, (int)SZ_X);

    attn_fwd<<<dim3(BATCH * NHEADS * (SEQ / 16)), dim3(32), 0, stream>>>(Qb, Kb, Vb, Yb);

    gemm_bf16_wmma<true><<<gg, gb, 0, stream>>>(Yb, wob, bo, nullptr, out, MROWS, D_MODEL, D_MODEL);
}
